// GNNEdgeAttr_72224170049680
// MI455X (gfx1250) — matrix operand order repaired and checked by reading, not compiled, not run on hardware
//
#include <hip/hip_runtime.h>
#include <hip/hip_bf16.h>

typedef __attribute__((ext_vector_type(16))) _Float16 v16h;
typedef __attribute__((ext_vector_type(2)))  _Float16 v2h;
typedef __attribute__((ext_vector_type(8)))  float    v8f;

#define N_NODES 10000
#define N_EDGES 160000

// ---------------- conversion kernels ----------------

__global__ void k_convert_pad(const float* __restrict__ in, _Float16* __restrict__ out,
                              int M, int F, int Kpad) {
  int idx = blockIdx.x * blockDim.x + threadIdx.x;
  if (idx >= M * Kpad) return;
  int m = idx / Kpad, kk = idx - m * Kpad;
  out[idx] = (kk < F) ? (_Float16)in[(size_t)m * F + kk] : (_Float16)0.f;
}

// W [K, Ncols] f32 -> WT [Ncols, Kpad] f16, zero-padded in K
__global__ void k_weight_T(const float* __restrict__ W, _Float16* __restrict__ WT,
                           int K, int Ncols, int Kpad) {
  int idx = blockIdx.x * blockDim.x + threadIdx.x;
  if (idx >= Ncols * Kpad) return;
  int n = idx / Kpad, kk = idx - n * Kpad;
  WT[idx] = (kk < K) ? (_Float16)W[(size_t)kk * Ncols + n] : (_Float16)0.f;
}

// ---- WMMA GEMM: out[M,Ncols] = A[M,Kpad](f16) * BT[Ncols,Kpad]^T(f16) + bias ----
// One wave per 16x64 output tile: one A fragment reused across 4 B fragments
// (4x v_wmma_f32_16x16x32_f16 per K-step; 40 B of L2 traffic per WMMA vs 64 B unblocked).

__global__ void k_wmma_gemm(const _Float16* __restrict__ A, const _Float16* __restrict__ BT,
                            const float* __restrict__ bias, float* __restrict__ out,
                            int M, int Kpad, int Ncols) {
  int wave = (blockIdx.x * blockDim.x + threadIdx.x) >> 5;
  int lane = threadIdx.x & 31;
  int tilesN4 = Ncols >> 6;                 // 64-wide column groups (Ncols % 64 == 0 here)
  int tiles   = (M >> 4) * tilesN4;
  if (wave >= tiles) return;                // wave-uniform: EXEC stays all-ones for WMMA
  int tm = wave / tilesN4, tn4 = wave - tm * tilesN4;
  int h2  = lane >> 4;
  int l15 = lane & 15;

  const _Float16* Arow  = A  + (size_t)(tm * 16 + l15) * Kpad;
  const _Float16* Bbase = BT + (size_t)(tn4 * 64 + l15) * Kpad + h2 * 16;
  const size_t bstride  = (size_t)16 * Kpad;   // halves between adjacent 16-col groups

  v8f acc0 = {}, acc1 = {}, acc2 = {}, acc3 = {};
  for (int k0 = 0; k0 < Kpad; k0 += 32) {
    v16h a;
#pragma unroll
    for (int j = 0; j < 8; ++j) {
      // A layout (contiguous, matches B): VGPR j holds K = h2*16 + 2*j (+0/+1)
      int kk = k0 + (h2 << 4) + (j << 1);
      v2h p = *(const v2h*)(Arow + kk);
      a[2 * j]     = p[0];
      a[2 * j + 1] = p[1];
    }
    // B layout: lane = N col; elements e=0..15 are K = h2*16 + e (contiguous, 32B aligned)
    v16h b0 = *(const v16h*)(Bbase + k0);
    v16h b1 = *(const v16h*)(Bbase + bstride + k0);
    v16h b2 = *(const v16h*)(Bbase + 2 * bstride + k0);
    v16h b3 = *(const v16h*)(Bbase + 3 * bstride + k0);
    acc0 = __builtin_amdgcn_wmma_f32_16x16x32_f16(false, a, false, b0, (short)0, acc0, false, false);
    acc1 = __builtin_amdgcn_wmma_f32_16x16x32_f16(false, a, false, b1, (short)0, acc1, false, false);
    acc2 = __builtin_amdgcn_wmma_f32_16x16x32_f16(false, a, false, b2, (short)0, acc2, false, false);
    acc3 = __builtin_amdgcn_wmma_f32_16x16x32_f16(false, a, false, b3, (short)0, acc3, false, false);
  }

  // C/D layout: VGPR g -> row (h2*8+g), col = lane&15
  int rbase = tm * 16 + h2 * 8;
  v8f accs[4] = {acc0, acc1, acc2, acc3};
#pragma unroll
  for (int t = 0; t < 4; ++t) {
    int col = tn4 * 64 + t * 16 + l15;
    float bb = bias ? bias[col] : 0.f;
#pragma unroll
    for (int g = 0; g < 8; ++g)
      out[(size_t)(rbase + g) * Ncols + col] = accs[t][g] + bb;
  }
}

// ---------------- attention kernels ----------------

__device__ inline unsigned f2ord(float f) {
  unsigned u = __float_as_uint(f);
  return (u & 0x80000000u) ? ~u : (u | 0x80000000u);
}
__device__ inline float ord2f(unsigned u) {
  return __uint_as_float((u & 0x80000000u) ? (u & 0x7FFFFFFFu) : ~u);
}

__global__ void k_init_ms(unsigned* __restrict__ menc, float* __restrict__ ssum, int n) {
  int i = blockIdx.x * blockDim.x + threadIdx.x;
  if (i < n) { menc[i] = 0u; ssum[i] = 0.f; }   // 0u orders below every finite logit
}

// logits[e,h] = <q[dst], k[src] + ea@We> / sqrt(C);  segment max via ordered-uint atomicMax
__global__ void k_edge_logits(const float* __restrict__ q, const float* __restrict__ kmat,
                              const int* __restrict__ src, const int* __restrict__ dst,
                              const float* __restrict__ ea, const float* __restrict__ We,
                              float* __restrict__ lg, unsigned* __restrict__ menc,
                              int E, int H, int C) {
  int idx = blockIdx.x * blockDim.x + threadIdx.x;
  if (idx >= E * H) return;
  int e = idx / H, h = idx - e * H;
  int s = src[e], d = dst[e];
  int HC = H * C;
  const float4* qd = (const float4*)(q    + (size_t)d * HC + h * C);
  const float4* ks = (const float4*)(kmat + (size_t)s * HC + h * C);
  const float4* w0 = (const float4*)(We + h * C);        // We row 0
  const float4* w1 = (const float4*)(We + HC + h * C);   // We row 1
  float a0 = ea[2 * e], a1 = ea[2 * e + 1];
  float acc = 0.f;
  int c4 = C >> 2;
#pragma unroll 4
  for (int c = 0; c < c4; ++c) {
    float4 qq = qd[c], kk = ks[c], e0 = w0[c], e1 = w1[c];
    acc += qq.x * (kk.x + a0 * e0.x + a1 * e1.x)
         + qq.y * (kk.y + a0 * e0.y + a1 * e1.y)
         + qq.z * (kk.z + a0 * e0.z + a1 * e1.z)
         + qq.w * (kk.w + a0 * e0.w + a1 * e1.w);
  }
  float lv = acc * rsqrtf((float)C);
  lg[idx] = lv;
  atomicMax(&menc[(size_t)d * H + h], f2ord(lv));
}

// p = exp(logit - m[dst]); segment sum via f32 atomics
__global__ void k_edge_p(float* __restrict__ lg, const int* __restrict__ dst,
                         const unsigned* __restrict__ menc, float* __restrict__ ssum,
                         int E, int H) {
  int idx = blockIdx.x * blockDim.x + threadIdx.x;
  if (idx >= E * H) return;
  int e = idx / H, h = idx - e * H;
  int d = dst[e];
  float m  = ord2f(menc[(size_t)d * H + h]);
  float pv = __expf(lg[idx] - m);
  lg[idx] = pv;
  atomicAdd(&ssum[(size_t)d * H + h], pv);
}

// out[dst] += (p/s[dst]) * (v[src] + ea@We); one thread per (edge, head, 4-wide C chunk)
__global__ void k_edge_agg(const float* __restrict__ vmat, const int* __restrict__ src,
                           const int* __restrict__ dst, const float* __restrict__ ea,
                           const float* __restrict__ We, const float* __restrict__ pbuf,
                           const float* __restrict__ ssum, float* __restrict__ out,
                           int E, int H, int C) {
  int c4n = C >> 2;
  int idx = blockIdx.x * blockDim.x + threadIdx.x;
  if (idx >= E * H * c4n) return;
  int eh = idx / c4n, c4 = idx - eh * c4n;
  int e = eh / H, h = eh - e * H;
  int s = src[e], d = dst[e];
  int HC = H * C;
  float alpha = pbuf[eh] / ssum[(size_t)d * H + h];
  int cb = c4 * 4;
  float4 vs = *(const float4*)(vmat + (size_t)s * HC + h * C + cb);
  float4 e0 = *(const float4*)(We + h * C + cb);
  float4 e1 = *(const float4*)(We + HC + h * C + cb);
  float a0 = ea[2 * e], a1 = ea[2 * e + 1];
  float* o = out + (size_t)d * HC + h * C + cb;
  atomicAdd(&o[0], alpha * (vs.x + a0 * e0.x + a1 * e1.x));
  atomicAdd(&o[1], alpha * (vs.y + a0 * e0.y + a1 * e1.y));
  atomicAdd(&o[2], alpha * (vs.z + a0 * e0.z + a1 * e1.z));
  atomicAdd(&o[3], alpha * (vs.w + a0 * e0.w + a1 * e1.w));
}

__global__ void k_relu(float* __restrict__ h, int n) {
  int i = blockIdx.x * blockDim.x + threadIdx.x;
  if (i < n) h[i] = fmaxf(h[i], 0.f);
}

// ---------------- host ----------------

extern "C" void kernel_launch(void* const* d_in, const int* in_sizes, int n_in,
                              void* d_out, int out_size, void* d_ws, size_t ws_size,
                              hipStream_t stream) {
  (void)in_sizes; (void)n_in; (void)out_size; (void)ws_size;
  const float* x  = (const float*)d_in[0];
  const int* esrc = (const int*)d_in[1];
  const int* edst = (const int*)d_in[2];
  const float* ea = (const float*)d_in[3];
  const float* W[3][9];
  for (int l = 0; l < 3; ++l)
    for (int j = 0; j < 9; ++j)
      W[l][j] = (const float*)d_in[4 + l * 9 + j];
  // per-layer param order: 0 Wq, 1 bq, 2 Wk, 3 bk, 4 Wv, 5 bv, 6 We, 7 Ws, 8 bs

  char* pw = (char*)d_ws;
  auto carve = [&](size_t bytes) { void* r = (void*)pw; pw += (bytes + 255) & ~(size_t)255; return r; };
  _Float16* h16   = (_Float16*)carve((size_t)N_NODES * 512 * 2);
  _Float16* wq16  = (_Float16*)carve((size_t)512 * 512 * 2);
  _Float16* wk16  = (_Float16*)carve((size_t)512 * 512 * 2);
  _Float16* wv16  = (_Float16*)carve((size_t)512 * 512 * 2);
  _Float16* ws16  = (_Float16*)carve((size_t)512 * 512 * 2);
  float*    qb    = (float*)carve((size_t)N_NODES * 512 * 4);
  float*    kb    = (float*)carve((size_t)N_NODES * 512 * 4);
  float*    vb    = (float*)carve((size_t)N_NODES * 512 * 4);
  float*    lgb   = (float*)carve((size_t)N_EDGES * 4 * 4);
  unsigned* menc  = (unsigned*)carve((size_t)N_NODES * 4 * 4);
  float*    ssum  = (float*)carve((size_t)N_NODES * 4 * 4);
  float*    hbuf1 = (float*)carve((size_t)N_NODES * 512 * 4);
  float*    hbuf2 = (float*)carve((size_t)N_NODES * 512 * 4);

  float* outf = (float*)d_out;

  struct Cfg { int fin, H, C; const float* hin; float* hout; bool relu; };
  Cfg cfg[3] = {
    {5,   4, 128, x,     hbuf1, true},
    {512, 4, 128, hbuf1, hbuf2, true},
    {512, 1, 64,  hbuf2, outf,  false},
  };

  const int M = N_NODES;
  for (int l = 0; l < 3; ++l) {
    int fin = cfg[l].fin, H = cfg[l].H, C = cfg[l].C;
    int Kpad = (fin + 31) & ~31;
    int HC = H * C;

    int tot = M * Kpad;
    k_convert_pad<<<(tot + 255) / 256, 256, 0, stream>>>(cfg[l].hin, h16, M, fin, Kpad);
    int wt = HC * Kpad;
    k_weight_T<<<(wt + 255) / 256, 256, 0, stream>>>(W[l][0], wq16, fin, HC, Kpad);
    k_weight_T<<<(wt + 255) / 256, 256, 0, stream>>>(W[l][2], wk16, fin, HC, Kpad);
    k_weight_T<<<(wt + 255) / 256, 256, 0, stream>>>(W[l][4], wv16, fin, HC, Kpad);
    k_weight_T<<<(wt + 255) / 256, 256, 0, stream>>>(W[l][7], ws16, fin, HC, Kpad);

    int tiles = (M / 16) * (HC / 64);   // one wave per 16x64 tile
    int gb = (tiles + 7) / 8;           // 8 waves / 256-thread block
    k_wmma_gemm<<<gb, 256, 0, stream>>>(h16, wq16, W[l][1], qb, M, Kpad, HC);
    k_wmma_gemm<<<gb, 256, 0, stream>>>(h16, wk16, W[l][3], kb, M, Kpad, HC);
    k_wmma_gemm<<<gb, 256, 0, stream>>>(h16, wv16, W[l][5], vb, M, Kpad, HC);
    k_wmma_gemm<<<gb, 256, 0, stream>>>(h16, ws16, W[l][8], cfg[l].hout, M, Kpad, HC); // skip -> out

    int nh = N_NODES * H;
    k_init_ms<<<(nh + 255) / 256, 256, 0, stream>>>(menc, ssum, nh);
    int eh = N_EDGES * H;
    k_edge_logits<<<(eh + 255) / 256, 256, 0, stream>>>(qb, kb, esrc, edst, ea, W[l][6],
                                                        lgb, menc, N_EDGES, H, C);
    k_edge_p<<<(eh + 255) / 256, 256, 0, stream>>>(lgb, edst, menc, ssum, N_EDGES, H);
    int agg = N_EDGES * H * (C / 4);
    k_edge_agg<<<(agg + 255) / 256, 256, 0, stream>>>(vb, esrc, edst, ea, W[l][6],
                                                      lgb, ssum, cfg[l].hout, N_EDGES, H, C);
    if (cfg[l].relu) {
      int t2 = M * HC;
      k_relu<<<(t2 + 255) / 256, 256, 0, stream>>>(cfg[l].hout, t2);
    }
  }
}
